// KANLinear_34050500723461
// MI455X (gfx1250) — compile-verified, matrix operand order checked by reading, not run on hardware
//
#include <hip/hip_runtime.h>

// KANLinear fused forward for MI455X (gfx1250, wave32, WMMA).
// out = silu(x) @ base_w^T + einsum('bic,oic->bo', bsplines(x), spline_w*scaler)
// One f16 WMMA GEMM over concatenated K = IN + IN*8 = 4608.
// v3: silu via v_rcp_f32 (no IEEE div chains); LDS double-buffering +
// one-slab-ahead register prefetch; float2 spline-x loads; 1 barrier/iter.

#define B_SZ   32768
#define IN_F   512
#define OUT_F  512
#define NCOEF  8
#define KTOT   (IN_F + IN_F * NCOEF)   // 4608
#define BM     128
#define BN     128
#define KC     32
#define LDT    40                      // padded LDS row stride (halves)

typedef __attribute__((ext_vector_type(16))) _Float16 v16h;
typedef __attribute__((ext_vector_type(8)))  _Float16 v8h;
typedef __attribute__((ext_vector_type(8)))  float    v8f;

__device__ __forceinline__ float fast_silu(float v) {
  // x * sigmoid(x); rcp is the hw v_rcp_f32 (~1 ulp) — ample for fp16 output.
  return v * __builtin_amdgcn_rcpf(1.0f + __expf(-v));
}

// ---------------------------------------------------------------------------
// Pack W (fp16, row-major [OUT][KTOT]).
// ---------------------------------------------------------------------------
__global__ __launch_bounds__(256) void kan_pack_w(
    const float* __restrict__ bw, const float* __restrict__ sw,
    const float* __restrict__ ss, _Float16* __restrict__ Wc) {
  int idx = blockIdx.x * 256 + threadIdx.x;
  if (idx >= OUT_F * KTOT) return;
  int o = idx / KTOT;
  int k = idx - o * KTOT;
  float w;
  if (k < IN_F) {
    w = bw[o * IN_F + k];
  } else {
    int t = k - IN_F;
    int i = t >> 3;
    int c = t & 7;
    w = sw[(o * IN_F + i) * NCOEF + c] * ss[o * IN_F + i];
  }
  Wc[idx] = (_Float16)w;
}

// ---------------------------------------------------------------------------
// 8 cubic B-spline basis values (uniform grid, closed form) packed to v8h.
// ---------------------------------------------------------------------------
__device__ __forceinline__ v8h bspline8(float xv, float g0, float invh) {
  float bas[NCOEF];
#pragma unroll
  for (int j = 0; j < NCOEF; ++j) bas[j] = 0.0f;
  float s   = (xv - g0) * invh;
  float fmf = floorf(s);
  int   mc  = (int)fmf;
  if (mc >= 0 && mc <= 10) {
    float u  = s - fmf;
    float um = 1.0f - u;
    float u2 = u * u, u3 = u2 * u;
    float nv[4] = {um * um * um * (1.0f / 6.0f),
                   (3.0f * u3 - 6.0f * u2 + 4.0f) * (1.0f / 6.0f),
                   (-3.0f * u3 + 3.0f * u2 + 3.0f * u + 1.0f) * (1.0f / 6.0f),
                   u3 * (1.0f / 6.0f)};
#pragma unroll
    for (int r = 0; r < 4; ++r) {
      int j = mc - 3 + r;
      if (j >= 0 && j < NCOEF) bas[j] = nv[r];
    }
  }
  v8h o;
#pragma unroll
  for (int c = 0; c < NCOEF; ++c) o[c] = (_Float16)bas[c];
  return o;
}

// ---------------------------------------------------------------------------
// Fragment loads (documented 16-bit A/B layout) + 2x4 WMMA tile update.
// ---------------------------------------------------------------------------
__device__ __forceinline__ void do_mma(const _Float16* __restrict__ a_t,
                                       const _Float16* __restrict__ b_t,
                                       int wm, int wn, int fm, int kb,
                                       v8f acc[2][4]) {
  v16h af[2];
#pragma unroll
  for (int a = 0; a < 2; ++a) {
    const _Float16* ap = a_t + (wm * 32 + a * 16 + fm) * LDT + kb;
    v8h lo = *(const v8h*)ap;          // K = kb .. kb+7
    v8h hi = *(const v8h*)(ap + 16);   // K = kb+16 .. kb+23
#pragma unroll
    for (int j = 0; j < 8; ++j) { af[a][j] = lo[j]; af[a][j + 8] = hi[j]; }
  }
  v16h bf[4];
#pragma unroll
  for (int b = 0; b < 4; ++b) {
    const _Float16* bp = b_t + (wn * 64 + b * 16 + fm) * LDT + kb;
    v8h lo = *(const v8h*)bp;
    v8h hi = *(const v8h*)(bp + 16);
#pragma unroll
    for (int j = 0; j < 8; ++j) { bf[b][j] = lo[j]; bf[b][j + 8] = hi[j]; }
  }
#pragma unroll
  for (int a = 0; a < 2; ++a)
#pragma unroll
    for (int b = 0; b < 4; ++b)
      acc[a][b] = __builtin_amdgcn_wmma_f32_16x16x32_f16(
          false, af[a], false, bf[b], (short)0, acc[a][b], false, false);
}

// ---------------------------------------------------------------------------
// Fused activation-gen + GEMM.  256 threads (8 waves), tile 128x128,
// waves 4(M) x 2(N), each wave 2x4 16x16 tiles, K stepped by 32.
// Double-buffered LDS, one-slab-ahead global prefetch, 1 barrier/iter.
// ---------------------------------------------------------------------------
__global__ __launch_bounds__(256) void kan_fused_gemm(
    const float* __restrict__ x, const _Float16* __restrict__ Wc,
    const float* __restrict__ grid, float* __restrict__ out) {
  __shared__ _Float16 aT[2][BM * LDT];
  __shared__ _Float16 bT[2][BN * LDT];
  __shared__ float    gk[12];

  const int t    = threadIdx.x;
  const int row0 = blockIdx.x * BM;
  const int col0 = blockIdx.y * BN;

  if (t < 12) gk[t] = grid[t];   // grid rows identical; row 0 suffices
  __syncthreads();
  const float g0   = gk[0];
  const float invh = __builtin_amdgcn_rcpf(gk[1] - gk[0]);

  const int wave = t >> 5;
  const int lane = t & 31;
  const int wm   = wave & 3;          // M quadrant
  const int wn   = wave >> 2;         // N half
  const int fm   = lane & 15;
  const int kb   = (lane >> 4) * 8;

  // staging assignments
  const int sn  = t >> 1;             // weight row (N) / silu row (M)
  const int shf = (t & 1) * 16;       // K half within slab
  const int sm  = t & 127;            // spline row (M)
  const int sjj = t >> 7;             // spline channel-pair selector (0/1)

  const _Float16* wsrc = Wc + (size_t)(col0 + sn) * KTOT + shf;
  const float*    xsil = x + (size_t)(row0 + sn) * IN_F + shf;
  const float*    xspl = x + (size_t)(row0 + sm) * IN_F + sjj * 2;

  v8h    pw0, pw1;                    // prefetched weight slab (16 halves)
  float4 px0, px1, px2, px3;          // prefetched silu x (16 floats)
  float2 sx;                          // prefetched spline x (2 channels)

  v8f acc[2][4] = {};
  int buf = 0;

  // ---- prologue prefetch (k0 = 0) ----
  pw0 = *(const v8h*)(wsrc);
  pw1 = *(const v8h*)(wsrc + 8);
  px0 = ((const float4*)xsil)[0];
  px1 = ((const float4*)xsil)[1];
  px2 = ((const float4*)xsil)[2];
  px3 = ((const float4*)xsil)[3];

  // =================== phase 1: silu(x) region, K = 0..511 ================
  for (int k0 = 0; k0 < IN_F; k0 += KC) {
    // stage prefetched regs -> LDS[buf]
    {
      _Float16* d = &bT[buf][sn * LDT + shf];
      *(v8h*)d       = pw0;
      *(v8h*)(d + 8) = pw1;
    }
    {
      _Float16* d = &aT[buf][sn * LDT + shf];
      float4 v[4] = {px0, px1, px2, px3};
#pragma unroll
      for (int j = 0; j < 4; ++j) {
        d[4 * j + 0] = (_Float16)fast_silu(v[j].x);
        d[4 * j + 1] = (_Float16)fast_silu(v[j].y);
        d[4 * j + 2] = (_Float16)fast_silu(v[j].z);
        d[4 * j + 3] = (_Float16)fast_silu(v[j].w);
      }
    }
    __syncthreads();

    // prefetch next slab while WMMAs run
    int kn = k0 + KC;
    pw0 = *(const v8h*)(wsrc + kn);
    pw1 = *(const v8h*)(wsrc + kn + 8);
    if (kn < IN_F) {
      const float* s = xsil + kn;
      px0 = ((const float4*)s)[0];
      px1 = ((const float4*)s)[1];
      px2 = ((const float4*)s)[2];
      px3 = ((const float4*)s)[3];
    } else {
      sx = *(const float2*)(xspl);   // first spline slab: i0 = 0
    }

    do_mma(aT[buf], bT[buf], wm, wn, fm, kb, acc);
    buf ^= 1;
  }

  // =============== phase 2: B-spline region, K = 512..4607 ================
  for (int k0 = IN_F; k0 < KTOT; k0 += KC) {
    {
      _Float16* d = &bT[buf][sn * LDT + shf];
      *(v8h*)d       = pw0;
      *(v8h*)(d + 8) = pw1;
    }
    {
      // this slab covers channels i0..i0+3; this thread: row sm,
      // channels 2*sjj and 2*sjj+1 (values prefetched in sx)
      _Float16* d = &aT[buf][sm * LDT + sjj * 16];
      *(v8h*)d       = bspline8(sx.x, g0, invh);
      *(v8h*)(d + 8) = bspline8(sx.y, g0, invh);
    }
    __syncthreads();

    int kn = k0 + KC;
    if (kn < KTOT) {
      pw0 = *(const v8h*)(wsrc + kn);
      pw1 = *(const v8h*)(wsrc + kn + 8);
      sx  = *(const float2*)(xspl + ((kn - IN_F) >> 3));
    }

    do_mma(aT[buf], bT[buf], wm, wn, fm, kb, acc);
    buf ^= 1;
  }

  // ---- epilogue: C/D layout N = lane&15, M = vgpr + 8*(lane>>4) ----
  const int n  = lane & 15;
  const int mb = (lane >> 4) * 8;
#pragma unroll
  for (int a = 0; a < 2; ++a) {
#pragma unroll
    for (int b = 0; b < 4; ++b) {
      int gcol = col0 + wn * 64 + b * 16 + n;
      int grow = row0 + wm * 32 + a * 16 + mb;
#pragma unroll
      for (int r = 0; r < 8; ++r)
        out[(size_t)(grow + r) * OUT_F + gcol] = acc[a][b][r];
    }
  }
}

// ---------------------------------------------------------------------------
extern "C" void kernel_launch(void* const* d_in, const int* in_sizes, int n_in,
                              void* d_out, int out_size, void* d_ws, size_t ws_size,
                              hipStream_t stream) {
  const float* x    = (const float*)d_in[0];   // (B, IN) f32
  const float* bw   = (const float*)d_in[1];   // (OUT, IN) f32
  const float* sw   = (const float*)d_in[2];   // (OUT, IN, 8) f32
  const float* ss   = (const float*)d_in[3];   // (OUT, IN) f32
  const float* grid = (const float*)d_in[4];   // (IN, 12) f32 (rows identical)
  float* out = (float*)d_out;                  // (B, OUT) f32

  _Float16* Wc = (_Float16*)d_ws;              // 512*4608*2 = 4.5 MB packed W

  kan_pack_w<<<(OUT_F * KTOT + 255) / 256, 256, 0, stream>>>(bw, sw, ss, Wc);

  dim3 g(B_SZ / BM, OUT_F / BN, 1);
  kan_fused_gemm<<<g, 256, 0, stream>>>(x, Wc, grid, out);
}
